// rgnn_75411035783753
// MI455X (gfx1250) — matrix operand order repaired and checked by reading, not compiled, not run on hardware
//
#include <hip/hip_runtime.h>
#include <hip/hip_bf16.h>
#include <math.h>

typedef __attribute__((ext_vector_type(16))) _Float16 v16h;
typedef __attribute__((ext_vector_type(8)))  float    v8f;

#define H   128
#define H2  256
#define H3  384
#define LIT 10
#define NCN 8000
#define NVN 16000
#define EE  60000
#define LN_EPS 1e-6f
#define IN_SCALE 0.1f

// ---------------------------------------------------------------------------
// WMMA fragment loaders (CDNA5 wave32 layouts, 16-bit operands)
// A 16x32 f16: lane<16 -> M=lane, K = {0..7, 16..23}; lane>=16 -> M=lane-16,
//              K = {8..15, 24..31}. Two contiguous 8-half groups per lane.
// B 32x16 f16: lane<16 -> N=lane, K = {0..7, 16..23}; lane>=16 -> N=lane-16, K = {8..15, 24..31}.
// C/D 16x16 f32: VGPR v -> M = v + 8*(lane>=16), N = lane%16.
// ---------------------------------------------------------------------------
__device__ __forceinline__ v16h load_a_frag(const _Float16* base, int lda,
                                            int kbase, int lane) {
  const int row  = lane & 15;
  const int koff = (lane & 16) ? 8 : 0;
  const _Float16* p = base + row * lda + kbase + koff;
  v16h a;
#pragma unroll
  for (int t = 0; t < 8; ++t) a[t] = p[t];
#pragma unroll
  for (int t = 0; t < 8; ++t) a[8 + t] = p[16 + t];
  return a;
}

__device__ __forceinline__ v16h load_b_frag(const _Float16* Bt, int ldk,
                                            int kbase, int colbase, int lane) {
  const int col  = colbase + (lane & 15);
  const int koff = (lane & 16) ? 8 : 0;
  const _Float16* p = Bt + (size_t)col * ldk + kbase + koff;
  v16h b;
#pragma unroll
  for (int t = 0; t < 8; ++t) b[t] = p[t];
#pragma unroll
  for (int t = 0; t < 8; ++t) b[8 + t] = p[16 + t];
  return b;
}

#define WMMA_F16(a, b, c) \
  __builtin_amdgcn_wmma_f32_16x16x32_f16(false, (a), false, (b), (short)0, (c), false, false)

__device__ __forceinline__ float elu_f(float x) {
  return (x > 0.f) ? x : (__expf(x) - 1.f);
}
__device__ __forceinline__ float sigmoid_f(float x) {
  return 1.f / (1.f + __expf(-x));
}

// ---------------------------------------------------------------------------
// Input transform + LayerNorm: h[n,j] = LN(0.1*f[n]*Win[j] + bin[j])
// one block (128 threads) per node
// ---------------------------------------------------------------------------
__global__ __launch_bounds__(128) void k_init_nodes(
    const float* __restrict__ feats, const float* __restrict__ Win,
    const float* __restrict__ bin, const float* __restrict__ g,
    const float* __restrict__ b, float* __restrict__ h,
    _Float16* __restrict__ h16) {
  __shared__ float ssum[128];
  __shared__ float ssq[128];
  const int node = blockIdx.x;
  const int j = threadIdx.x;
  const float x = IN_SCALE * feats[node] * Win[j] + bin[j];
  ssum[j] = x;
  ssq[j]  = x * x;
  __syncthreads();
#pragma unroll
  for (int s = 64; s > 0; s >>= 1) {
    if (j < s) { ssum[j] += ssum[j + s]; ssq[j] += ssq[j + s]; }
    __syncthreads();
  }
  const float mean = ssum[0] * (1.f / H);
  const float var  = ssq[0] * (1.f / H) - mean * mean;
  const float y = g[j] * (x - mean) * rsqrtf(var + LN_EPS) + b[j];
  const size_t idx = (size_t)node * H + j;
  h[idx]   = y;
  h16[idx] = (_Float16)y;
}

// ---------------------------------------------------------------------------
// Weight convert + transpose: W[K][N] f32 -> Bt[N][K] f16
// ---------------------------------------------------------------------------
__global__ void k_w_transpose(const float* __restrict__ W,
                              _Float16* __restrict__ Bt, int K, int N) {
  int idx = blockIdx.x * blockDim.x + threadIdx.x;
  if (idx >= K * N) return;
  const int k = idx / N, n = idx % N;
  Bt[(size_t)n * K + k] = (_Float16)W[idx];
}

// ---------------------------------------------------------------------------
// Edge messages: m = ELU(concat(h_src[esrc], h_tgt[etgt]) @ Wmsg + b)
// then scatter-sum into agg[etgt]. One block = 16 edges, 8 waves cover 8
// column tiles of H=128. K = 2H = 256 -> 8 WMMAs per wave.
// ---------------------------------------------------------------------------
__global__ __launch_bounds__(256) void k_edge_msg(
    const _Float16* __restrict__ hsrc, const _Float16* __restrict__ htgt,
    const int* __restrict__ esrc, const int* __restrict__ etgt,
    const _Float16* __restrict__ Wt, const float* __restrict__ bias,
    float* __restrict__ agg) {
  __shared__ __attribute__((aligned(16))) _Float16 Als[16 * H2];
  __shared__ int tg[16];
  const int e0 = blockIdx.x * 16;
  const int t = threadIdx.x;
  {
    const int r = t >> 4, c = t & 15;
    const int e = e0 + r;
    const _Float16* sp = (c < 8)
        ? (hsrc + (size_t)esrc[e] * H + c * 16)
        : (htgt + (size_t)etgt[e] * H + (c - 8) * 16);
    const uint4* s4 = reinterpret_cast<const uint4*>(sp);
    uint4* d4 = reinterpret_cast<uint4*>(Als + r * H2 + c * 16);
    d4[0] = s4[0];
    d4[1] = s4[1];
    if (t < 16) tg[t] = etgt[e0 + t];
  }
  __syncthreads();
  const int lane = t & 31;
  const int n0 = (t >> 5) * 16;
  v8f acc = {};
#pragma unroll
  for (int ks = 0; ks < 8; ++ks) {
    v16h a = load_a_frag(Als, H2, ks * 32, lane);
    v16h b = load_b_frag(Wt, H2, ks * 32, n0, lane);
    acc = WMMA_F16(a, b, acc);
  }
  const int col = n0 + (lane & 15);
  const float bb = bias[col];
  const int mo = (lane & 16) ? 8 : 0;
#pragma unroll
  for (int v = 0; v < 8; ++v) {
    const float x = elu_f(acc[v] + bb);
    atomicAdd(agg + (size_t)tg[v + mo] * H + col, x);
  }
}

// ---------------------------------------------------------------------------
// Fused node update:
//   upd = ELU(concat(h, agg) @ Wns + bns)            (phase B, K=256)
//   xa  = upd @ Wg + bg0 ; ha = h @ Ug + bg1         (phase C, K=128 each)
//   z = sig(xa_z+ha_z); r = sig(xa_r+ha_r)
//   hh = tanh(xa_c + r*ha_c); h = z*h + (1-z)*hh
// One block = 16 nodes; 8 waves. z/r gates accumulate both GEMMs into one C.
// ---------------------------------------------------------------------------
__global__ __launch_bounds__(256) void k_node_update(
    float* __restrict__ h, _Float16* __restrict__ h16,
    const float* __restrict__ agg,
    const _Float16* __restrict__ WnsT, const float* __restrict__ bns,
    const _Float16* __restrict__ WgT, const _Float16* __restrict__ UgT,
    const float* __restrict__ bg) {
  __shared__ __attribute__((aligned(16))) _Float16 Als[16 * H2];  // [h | agg]
  __shared__ __attribute__((aligned(16))) _Float16 Uls[16 * H];   // upd f16
  const int n0 = blockIdx.x * 16;
  const int t = threadIdx.x;
  {
    const int r = t >> 4, c = t & 15;
    const size_t node = n0 + r;
    if (c < 8) {
      const uint4* s4 = reinterpret_cast<const uint4*>(h16 + node * H + c * 16);
      uint4* d4 = reinterpret_cast<uint4*>(Als + r * H2 + c * 16);
      d4[0] = s4[0];
      d4[1] = s4[1];
    } else {
      const float* ap = agg + node * H + (c - 8) * 16;
      _Float16* dp = Als + r * H2 + 128 + (c - 8) * 16;
#pragma unroll
      for (int i = 0; i < 16; ++i) dp[i] = (_Float16)ap[i];
    }
  }
  __syncthreads();
  const int lane = t & 31;
  const int wave = t >> 5;
  const int laneN = lane & 15;
  const int mo = (lane & 16) ? 8 : 0;
  // ---- phase B: upd tile, cols [16*wave, 16*wave+16)
  {
    const int c0 = wave * 16;
    v8f acc = {};
#pragma unroll
    for (int ks = 0; ks < 8; ++ks) {
      v16h a = load_a_frag(Als, H2, ks * 32, lane);
      v16h b = load_b_frag(WnsT, H2, ks * 32, c0, lane);
      acc = WMMA_F16(a, b, acc);
    }
    const int col = c0 + laneN;
    const float bb = bns[col];
#pragma unroll
    for (int v = 0; v < 8; ++v) {
      Uls[(v + mo) * H + col] = (_Float16)elu_f(acc[v] + bb);
    }
  }
  __syncthreads();
  // ---- phase C: GRU, gate-local cols [16*wave, 16*wave+16)
  {
    const int gc = wave * 16;
    v8f az = {}, ar = {}, axc = {}, ahc = {};
#pragma unroll
    for (int ks = 0; ks < 4; ++ks) {
      const int kb = ks * 32;
      v16h au = load_a_frag(Uls, H, kb, lane);   // upd (GRU input x)
      v16h ah = load_a_frag(Als, H2, kb, lane);  // old state h (first 128 cols)
      az  = WMMA_F16(au, load_b_frag(WgT, H, kb, gc, lane), az);
      az  = WMMA_F16(ah, load_b_frag(UgT, H, kb, gc, lane), az);
      ar  = WMMA_F16(au, load_b_frag(WgT, H, kb, H + gc, lane), ar);
      ar  = WMMA_F16(ah, load_b_frag(UgT, H, kb, H + gc, lane), ar);
      axc = WMMA_F16(au, load_b_frag(WgT, H, kb, 2 * H + gc, lane), axc);
      ahc = WMMA_F16(ah, load_b_frag(UgT, H, kb, 2 * H + gc, lane), ahc);
    }
    const int j = gc + laneN;
    const float bz  = bg[j] + bg[H3 + j];
    const float br  = bg[H + j] + bg[H3 + H + j];
    const float bxc = bg[2 * H + j];
    const float bhc = bg[H3 + 2 * H + j];
#pragma unroll
    for (int v = 0; v < 8; ++v) {
      const size_t idx = (size_t)(n0 + v + mo) * H + j;
      const float z  = sigmoid_f(az[v] + bz);
      const float r  = sigmoid_f(ar[v] + br);
      const float hh = tanhf(axc[v] + bxc + r * (ahc[v] + bhc));
      const float hold = h[idx];
      const float hnew = z * hold + (1.f - z) * hh;
      h[idx]   = hnew;
      h16[idx] = (_Float16)hnew;
    }
  }
}

// ---------------------------------------------------------------------------
// Final predictor: out[n] = dot(hv[n,:], Wf) + bf
// ---------------------------------------------------------------------------
__global__ __launch_bounds__(128) void k_final(const float* __restrict__ hv,
                                               const float* __restrict__ Wf,
                                               const float* __restrict__ bf,
                                               float* __restrict__ out) {
  __shared__ float red[128];
  const int node = blockIdx.x;
  const int j = threadIdx.x;
  red[j] = hv[(size_t)node * H + j] * Wf[j];
  __syncthreads();
#pragma unroll
  for (int s = 64; s > 0; s >>= 1) {
    if (j < s) red[j] += red[j + s];
    __syncthreads();
  }
  if (j == 0) out[node] = red[0] + bf[0];
}

// ---------------------------------------------------------------------------
extern "C" void kernel_launch(void* const* d_in, const int* in_sizes, int n_in,
                              void* d_out, int out_size, void* d_ws,
                              size_t ws_size, hipStream_t stream) {
  (void)in_sizes; (void)n_in; (void)out_size; (void)ws_size;
  const float* check_feats = (const float*)d_in[0];
  const float* var_feats   = (const float*)d_in[1];
  const int*   c2v_src     = (const int*)d_in[2];
  const int*   c2v_tgt     = (const int*)d_in[3];
  const int*   v2c_src     = (const int*)d_in[4];
  const int*   v2c_tgt     = (const int*)d_in[5];
  const float* Wc_in = (const float*)d_in[6];
  const float* bc_in = (const float*)d_in[7];
  const float* Wv_in = (const float*)d_in[8];
  const float* bv_in = (const float*)d_in[9];
  const float* gc_ln = (const float*)d_in[10];
  const float* bc_ln = (const float*)d_in[11];
  const float* gv_ln = (const float*)d_in[12];
  const float* bv_ln = (const float*)d_in[13];
  const float* Wmsg_c = (const float*)d_in[14];
  const float* bmsg_c = (const float*)d_in[15];
  const float* Wmsg_v = (const float*)d_in[16];
  const float* bmsg_v = (const float*)d_in[17];
  const float* Wns_c = (const float*)d_in[18];
  const float* bns_c = (const float*)d_in[19];
  const float* Wns_v = (const float*)d_in[20];
  const float* bns_v = (const float*)d_in[21];
  const float* Wg_c = (const float*)d_in[22];
  const float* Ug_c = (const float*)d_in[23];
  const float* bg_c = (const float*)d_in[24];
  const float* Wg_v = (const float*)d_in[25];
  const float* Ug_v = (const float*)d_in[26];
  const float* bg_v = (const float*)d_in[27];
  const float* Wf = (const float*)d_in[28];
  const float* bf = (const float*)d_in[29];
  float* out = (float*)d_out;

  // ---- carve workspace
  char* ws = (char*)d_ws;
  size_t off = 0;
  auto carve = [&](size_t bytes) -> void* {
    void* p = ws + off;
    off += (bytes + 255) & ~(size_t)255;
    return p;
  };
  float*     hc     = (float*)carve((size_t)NCN * H * 4);
  float*     hv     = (float*)carve((size_t)NVN * H * 4);
  float*     agg_c  = (float*)carve((size_t)NCN * H * 4);
  float*     agg_v  = (float*)carve((size_t)NVN * H * 4);
  _Float16*  hc16   = (_Float16*)carve((size_t)NCN * H * 2);
  _Float16*  hv16   = (_Float16*)carve((size_t)NVN * H * 2);
  _Float16*  WmsgTc = (_Float16*)carve((size_t)H2 * H * 2);
  _Float16*  WmsgTv = (_Float16*)carve((size_t)H2 * H * 2);
  _Float16*  WnsTc  = (_Float16*)carve((size_t)H2 * H * 2);
  _Float16*  WnsTv  = (_Float16*)carve((size_t)H2 * H * 2);
  _Float16*  WgTc   = (_Float16*)carve((size_t)H * H3 * 2);
  _Float16*  UgTc   = (_Float16*)carve((size_t)H * H3 * 2);
  _Float16*  WgTv   = (_Float16*)carve((size_t)H * H3 * 2);
  _Float16*  UgTv   = (_Float16*)carve((size_t)H * H3 * 2);

  // ---- weight convert+transpose (recomputed every call: deterministic)
  auto xpose = [&](const float* W, _Float16* Bt, int K, int N) {
    const int total = K * N;
    k_w_transpose<<<(total + 255) / 256, 256, 0, stream>>>(W, Bt, K, N);
  };
  xpose(Wmsg_c, WmsgTc, H2, H);
  xpose(Wmsg_v, WmsgTv, H2, H);
  xpose(Wns_c,  WnsTc,  H2, H);
  xpose(Wns_v,  WnsTv,  H2, H);
  xpose(Wg_c,   WgTc,   H,  H3);
  xpose(Ug_c,   UgTc,   H,  H3);
  xpose(Wg_v,   WgTv,   H,  H3);
  xpose(Ug_v,   UgTv,   H,  H3);

  // ---- input transform + LayerNorm
  k_init_nodes<<<NCN, 128, 0, stream>>>(check_feats, Wc_in, bc_in, gc_ln, bc_ln,
                                        hc, hc16);
  k_init_nodes<<<NVN, 128, 0, stream>>>(var_feats, Wv_in, bv_in, gv_ln, bv_ln,
                                        hv, hv16);

  // ---- message-passing iterations
  for (int it = 0; it < LIT; ++it) {
    hipMemsetAsync(agg_c, 0, (size_t)NCN * H * 4, stream);
    hipMemsetAsync(agg_v, 0, (size_t)NVN * H * 4, stream);
    // v -> c : concat(hv[src], hc[tgt]) @ Wmsg_c, scatter into agg_c
    k_edge_msg<<<EE / 16, 256, 0, stream>>>(hv16, hc16, v2c_src, v2c_tgt,
                                            WmsgTc, bmsg_c, agg_c);
    // c -> v : concat(hc[src], hv[tgt]) @ Wmsg_v, scatter into agg_v
    k_edge_msg<<<EE / 16, 256, 0, stream>>>(hc16, hv16, c2v_src, c2v_tgt,
                                            WmsgTv, bmsg_v, agg_v);
    // node updates (fused NextState + GRU)
    k_node_update<<<NCN / 16, 256, 0, stream>>>(hc, hc16, agg_c, WnsTc, bns_c,
                                                WgTc, UgTc, bg_c);
    k_node_update<<<NVN / 16, 256, 0, stream>>>(hv, hv16, agg_v, WnsTv, bns_v,
                                                WgTv, UgTv, bg_v);
  }

  // ---- final predictor
  k_final<<<NVN, 128, 0, stream>>>(hv, Wf, bf, out);
}
